// GRSA_23192823398635
// MI455X (gfx1250) — matrix operand order repaired and checked by reading, not compiled, not run on hardware
//
#include <hip/hip_runtime.h>
#include <hip/hip_bf16.h>

typedef __attribute__((ext_vector_type(16))) _Float16 v16h;
typedef __attribute__((ext_vector_type(8)))  _Float16 v8h;
typedef __attribute__((ext_vector_type(4)))  _Float16 v4h;
typedef __attribute__((ext_vector_type(8)))  float    v8f;

#define DIMC   192
#define NTOK   64
#define NHEAD  6
#define HALFC  96
#define LOG100 4.60517018598809136804f

// ---- LDS layout (bytes), phase-overlaid -------------------------------------
#define OFF_XH    0         // x f16 [64][192]            (dead after QKV)
#define OFF_KT    24576     // k unnormalized f16 [64][192] (dead after norm)
#define OFF_QH    49152     // q f16 [64][192] (normalized in place)
#define OFF_KSWZ  73728     // k^T swizzled B-frags: [6][4 mtile][32 lane][16]
#define OFF_VSWZ  98304     // v swizzled B-frags: [6][2 kstep][2 dtile][32][16]
#define OFF_SC    122880    // scores f32 [6][64][64]     (dead after softmax)
#define OFF_PATTN 0         // probs f16 [6][64][64]      overlays XH+KT
#define OFF_O     122880    // attn-out f16 [64][192]     overlays scores
#define SMEM_BYTES (122880 + 98304)   // 216 KB < 320 KB/WGP

// ---- WMMA fragment helpers (16-bit operands, contiguous k = 16*(lane/16) + s) ---------------
__device__ __forceinline__ v8f wmma16(v16h a, v16h b, v8f c) {
    return __builtin_amdgcn_wmma_f32_16x16x32_f16(false, a, false, b,
                                                  (short)0, c, false, false);
}

// A (16x32 f16, row-major src, leading dim ldm)
__device__ __forceinline__ v16h load_a_frag(const _Float16* src, int ldm) {
    const int lane = threadIdx.x & 31;
    const _Float16* p = src + (lane & 15) * ldm + ((lane >> 4) << 4);
    v8h lo = *(const v8h*)p;
    v8h hi = *(const v8h*)(p + 8);
    v16h a;
#pragma unroll
    for (int i = 0; i < 8; ++i) { a[i] = lo[i]; a[i + 8] = hi[i]; }
    return a;
}

// B (32x16 f16) from pre-swizzled [lane][16] storage
__device__ __forceinline__ v16h load_b_frag(const _Float16* base) {
    const int lane = threadIdx.x & 31;
    const _Float16* p = base + lane * 16;
    v8h lo = *(const v8h*)p;
    v8h hi = *(const v8h*)(p + 8);
    v16h b;
#pragma unroll
    for (int i = 0; i < 8; ++i) { b[i] = lo[i]; b[i + 8] = hi[i]; }
    return b;
}

__device__ __forceinline__ v8f zero8() {
    v8f z;
#pragma unroll
    for (int i = 0; i < 8; ++i) z[i] = 0.f;
    return z;
}

// ---- pre-pass 1: swizzle 8 weight matrices into B-fragment order ------------
__global__ void grsa_swz_kernel(const float* __restrict__ w0, const float* __restrict__ w1,
                                const float* __restrict__ w2, const float* __restrict__ w3,
                                const float* __restrict__ w4, const float* __restrict__ w5,
                                const float* __restrict__ w6, const float* __restrict__ w7,
                                _Float16* __restrict__ wswz) {
    int i = blockIdx.x * 256 + threadIdx.x;
    if (i >= 8 * 9216) return;
    int mi = i / 9216, rem = i % 9216;
    int kt = rem / 3072, rem2 = rem % 3072;
    int nt = rem2 / 512, li = rem2 % 512;
    int lane = li >> 4, j = li & 15;
    int k = kt * 32 + ((lane >> 4) << 4) + j;   // input channel
    int n = nt * 16 + (lane & 15);              // output channel
    const float* W = (mi == 0) ? w0 : (mi == 1) ? w1 : (mi == 2) ? w2 :
                     (mi == 3) ? w3 : (mi == 4) ? w4 : (mi == 5) ? w5 :
                     (mi == 6) ? w6 : w7;
    wswz[i] = (_Float16)W[n * 96 + k];
}

// ---- pre-pass 2: rpb MLP -> 16*sigmoid bias table [6][64][64] f32 -----------
__global__ void grsa_bias_kernel(const float* __restrict__ tabin,
                                 const float* __restrict__ w1,
                                 const float* __restrict__ b1,
                                 const float* __restrict__ w2,
                                 const int*   __restrict__ idx,
                                 float* __restrict__ biasPlus) {
    __shared__ float tab[225 * 6];
    int tid = threadIdx.x;
    if (tid < 225) {
        float t0 = tabin[tid * 2 + 0], t1 = tabin[tid * 2 + 1];
        float o[6] = {0.f, 0.f, 0.f, 0.f, 0.f, 0.f};
        for (int j = 0; j < 128; ++j) {
            float hj = fmaxf(t0 * w1[2 * j] + t1 * w1[2 * j + 1] + b1[j], 0.f);
#pragma unroll
            for (int h = 0; h < 6; ++h) o[h] += w2[h * 128 + j] * hj;
        }
#pragma unroll
        for (int h = 0; h < 6; ++h) tab[tid * 6 + h] = o[h];
    }
    __syncthreads();
    for (int e = tid; e < 6 * 4096; e += 256) {
        int h = e >> 12, nm = e & 4095;
        float v = tab[idx[nm] * 6 + h];
        biasPlus[e] = 16.f / (1.f + __expf(-v));
    }
}

// ---- main fused window-attention kernel: 1 block (4 waves) per window -------
__global__ __launch_bounds__(128) void grsa_main(
    const float* __restrict__ x, const float* __restrict__ mask,
    const float* __restrict__ ls, const _Float16* __restrict__ wswz,
    const float* __restrict__ biasPlus,
    const float* __restrict__ q1b, const float* __restrict__ q2b,
    const float* __restrict__ k1b, const float* __restrict__ k2b,
    const float* __restrict__ v1b, const float* __restrict__ v2b,
    const float* __restrict__ p1b, const float* __restrict__ p2b,
    float* __restrict__ out) {
    extern __shared__ __align__(16) char smem[];
    _Float16* sm_x    = (_Float16*)(smem + OFF_XH);
    _Float16* sm_kt   = (_Float16*)(smem + OFF_KT);
    _Float16* sm_q    = (_Float16*)(smem + OFF_QH);
    _Float16* sm_kswz = (_Float16*)(smem + OFF_KSWZ);
    _Float16* sm_vswz = (_Float16*)(smem + OFF_VSWZ);
    float*    sm_sc   = (float*)   (smem + OFF_SC);
    _Float16* sm_p    = (_Float16*)(smem + OFF_PATTN);
    _Float16* sm_o    = (_Float16*)(smem + OFF_O);

    const int tid  = threadIdx.x;
    // wave id is wave-uniform by construction; readfirstlane makes the
    // compiler treat all tile-decode arithmetic as scalar (SGPR) work.
    const int wave = __builtin_amdgcn_readfirstlane(tid >> 5);
    const int lane = tid & 31;
    const int colN = lane & 15;
    const int rowB = (lane >> 4) << 3;
    const int bid  = blockIdx.x;
    const int win  = bid & 63;               // mask window index (b % NW)

    // ---- phase 0: stage x[bid] (64x192 f32) -> LDS f16 ----------------------
    const float* xg = x + (size_t)bid * (NTOK * DIMC);
    __builtin_prefetch(xg + tid * 96, 0, 0);          // global_prefetch_b8
    const float4* xg4 = (const float4*)xg;
#pragma unroll 4
    for (int i = tid; i < (NTOK * DIMC) / 4; i += 128) {
        float4 v = xg4[i];
        v4h hv; hv[0] = (_Float16)v.x; hv[1] = (_Float16)v.y;
                hv[2] = (_Float16)v.z; hv[3] = (_Float16)v.w;
        *(v4h*)(sm_x + i * 4) = hv;
    }
    __syncthreads();

    // ---- phase 1: QKV projections: y = xh @ W^T + b + xh --------------------
    // proj is compile-time (unrolled) -> store routing & bias select are static
#pragma unroll
    for (int proj = 0; proj < 3; ++proj) {
        const float* blo = (proj == 0) ? q1b : (proj == 1) ? k1b : v1b;
        const float* bhi = (proj == 0) ? q2b : (proj == 1) ? k2b : v2b;
        for (int t = wave; t < 48; t += 4) {
            int half = t / 24, idx = t % 24;
            int mt = idx / 6, nt = idx % 6;
            v8f acc = zero8();
#pragma unroll
            for (int kt = 0; kt < 3; ++kt) {
                v16h a = load_a_frag(sm_x + mt * 16 * DIMC + half * HALFC + kt * 32, DIMC);
                v16h b = load_b_frag(wswz + (proj * 2 + half) * 9216 + (kt * 6 + nt) * 512);
                acc = wmma16(a, b, acc);
            }
            const float* bias = half ? bhi : blo;
            float bv = bias[nt * 16 + colN];
#pragma unroll
            for (int r = 0; r < 8; ++r) {
                int m  = mt * 16 + rowB + r;
                int ch = half * HALFC + nt * 16 + colN;
                float val = acc[r] + bv + (float)sm_x[m * DIMC + ch];
                if (proj == 0) {
                    sm_q[m * DIMC + ch] = (_Float16)val;
                } else if (proj == 1) {
                    sm_kt[m * DIMC + ch] = (_Float16)val;
                } else { // V: scatter directly into B-fragment order for P@V
                    int h = ch >> 5, d = ch & 31;
                    int ks = m >> 5, ntd = d >> 4;
                    int lp = (((m & 31) >> 4) << 4) + (d & 15);
                    sm_vswz[(((h * 2 + ks) * 2 + ntd) * 32 + lp) * 16 + (m & 15)] =
                        (_Float16)val;
                }
            }
        }
    }
    __syncthreads();

    // ---- phase 2: per-(token,head) L2 norm; fold logit scale into q ---------
    for (int r = tid; r < NTOK * NHEAD; r += 128) {
        int h = r >> 6, n = r & 63;
        float scale = __expf(fminf(ls[h], LOG100));
        _Float16* qr = sm_q  + n * DIMC + h * 32;
        _Float16* kr = sm_kt + n * DIMC + h * 32;
        float sq = 0.f, sk = 0.f;
#pragma unroll
        for (int d = 0; d < 32; ++d) {
            float a = (float)qr[d]; sq += a * a;
            float b = (float)kr[d]; sk += b * b;
        }
        float invq = scale / fmaxf(sqrtf(sq), 1e-12f);
        float invk = 1.f   / fmaxf(sqrtf(sk), 1e-12f);
#pragma unroll
        for (int d = 0; d < 32; ++d) {
            qr[d] = (_Float16)((float)qr[d] * invq);
            float kv = (float)kr[d] * invk;
            int lp = ((d >> 4) << 4) + (n & 15);
            sm_kswz[((h * 4 + (n >> 4)) * 32 + lp) * 16 + (d & 15)] = (_Float16)kv;
        }
    }
    __syncthreads();

    // ---- phase 3: scores S = qn @ kn^T  (+ rpb bias + mask) -----------------
    for (int t = wave; t < 96; t += 4) {
        int h = t >> 4, rem = t & 15;
        int mt = rem >> 2, nts = rem & 3;
        v16h a = load_a_frag(sm_q + mt * 16 * DIMC + h * 32, DIMC);
        v16h b = load_b_frag(sm_kswz + (h * 4 + nts) * 512);
        v8f acc = wmma16(a, b, zero8());
#pragma unroll
        for (int r = 0; r < 8; ++r) {
            int n = mt * 16 + rowB + r, m = nts * 16 + colN;
            float v = acc[r] + biasPlus[(h * 64 + n) * 64 + m]
                             + mask[((size_t)win * 64 + n) * 64 + m];
            sm_sc[(h * 64 + n) * 64 + m] = v;
        }
    }
    __syncthreads();

    // ---- phase 4: row softmax (384 rows of 64), single exp pass -------------
    for (int r = tid; r < NTOK * NHEAD; r += 128) {
        int h = r >> 6, n = r & 63;
        const float* row = sm_sc + (h * 64 + n) * 64;
        _Float16* prow = sm_p + (h * 64 + n) * 64;
        float mx = -3.0e38f;
#pragma unroll 8
        for (int m = 0; m < 64; ++m) mx = fmaxf(mx, row[m]);
        float s = 0.f;
#pragma unroll 8
        for (int m = 0; m < 64; ++m) {
            float e = __expf(row[m] - mx);
            s += e;
            prow[m] = (_Float16)e;
        }
        float inv = 1.f / s;
#pragma unroll 8
        for (int m = 0; m < 64; ++m)
            prow[m] = (_Float16)((float)prow[m] * inv);
    }
    __syncthreads();

    // ---- phase 5: O = P @ V  (6 heads x 4 mtiles x 2 dtiles, K=64) ----------
    for (int t = wave; t < 48; t += 4) {
        int h = t / 8, rem = t % 8;
        int mt = rem >> 1, ntd = rem & 1;
        v8f acc = zero8();
#pragma unroll
        for (int ks = 0; ks < 2; ++ks) {
            v16h a = load_a_frag(sm_p + h * 4096 + mt * 16 * 64 + ks * 32, 64);
            v16h b = load_b_frag(sm_vswz + ((h * 2 + ks) * 2 + ntd) * 512);
            acc = wmma16(a, b, acc);
        }
#pragma unroll
        for (int r = 0; r < 8; ++r) {
            int n = mt * 16 + rowB + r;
            int c = h * 32 + ntd * 16 + colN;
            sm_o[n * DIMC + c] = (_Float16)acc[r];
        }
    }
    __syncthreads();

    // ---- phase 6: output projection (no residual), f32 store ----------------
    float* og = out + (size_t)bid * (NTOK * DIMC);
    for (int t = wave; t < 48; t += 4) {
        int half = t / 24, idx = t % 24;
        int mt = idx / 6, nt = idx % 6;
        v8f acc = zero8();
#pragma unroll
        for (int kt = 0; kt < 3; ++kt) {
            v16h a = load_a_frag(sm_o + mt * 16 * DIMC + half * HALFC + kt * 32, DIMC);
            v16h b = load_b_frag(wswz + (6 + half) * 9216 + (kt * 6 + nt) * 512);
            acc = wmma16(a, b, acc);
        }
        const float* pb = half ? p2b : p1b;
        float bv = pb[nt * 16 + colN];
#pragma unroll
        for (int r = 0; r < 8; ++r) {
            int n  = mt * 16 + rowB + r;
            int ch = half * HALFC + nt * 16 + colN;
            og[n * DIMC + ch] = acc[r] + bv;
        }
    }
}

extern "C" void kernel_launch(void* const* d_in, const int* in_sizes, int n_in,
                              void* d_out, int out_size, void* d_ws, size_t ws_size,
                              hipStream_t stream) {
    (void)in_sizes; (void)n_in; (void)out_size; (void)ws_size;
    const float* x        = (const float*)d_in[0];
    const float* mask     = (const float*)d_in[1];
    const float* ls       = (const float*)d_in[2];
    const float* mlp_w1   = (const float*)d_in[3];
    const float* mlp_b1   = (const float*)d_in[4];
    const float* mlp_w2   = (const float*)d_in[5];
    const float* q1w = (const float*)d_in[6];  const float* q1b = (const float*)d_in[7];
    const float* q2w = (const float*)d_in[8];  const float* q2b = (const float*)d_in[9];
    const float* k1w = (const float*)d_in[10]; const float* k1b = (const float*)d_in[11];
    const float* k2w = (const float*)d_in[12]; const float* k2b = (const float*)d_in[13];
    const float* v1w = (const float*)d_in[14]; const float* v1b = (const float*)d_in[15];
    const float* v2w = (const float*)d_in[16]; const float* v2b = (const float*)d_in[17];
    const float* p1w = (const float*)d_in[18]; const float* p1b = (const float*)d_in[19];
    const float* p2w = (const float*)d_in[20]; const float* p2b = (const float*)d_in[21];
    const float* rpb_table = (const float*)d_in[22];
    const int*   rp_index  = (const int*)d_in[23];
    float* out = (float*)d_out;

    _Float16* wswz   = (_Float16*)d_ws;                       // 8*9216 f16 = 144 KB
    float* biasPlus  = (float*)((char*)d_ws + 8 * 9216 * 2);  // 6*64*64 f32 = 96 KB

    grsa_swz_kernel<<<(8 * 9216 + 255) / 256, 256, 0, stream>>>(
        q1w, q2w, k1w, k2w, v1w, v2w, p1w, p2w, wswz);
    grsa_bias_kernel<<<1, 256, 0, stream>>>(
        rpb_table, mlp_w1, mlp_b1, mlp_w2, rp_index, biasPlus);

    (void)hipFuncSetAttribute((const void*)grsa_main,
                              hipFuncAttributeMaxDynamicSharedMemorySize,
                              SMEM_BYTES);
    grsa_main<<<4096, 128, SMEM_BYTES, stream>>>(
        x, mask, ls, wswz, biasPlus,
        q1b, q2b, k1b, k2b, v1b, v2b, p1b, p2b, out);
}
